// Net_3135326126715
// MI455X (gfx1250) — matrix operand order repaired and checked by reading, not compiled, not run on hardware
//
#include <hip/hip_runtime.h>
#include <hip/hip_bf16.h>

// =====================================================================
// MoNet / GMMConv GNN forward for MI455X (gfx1250, wave32, WMMA).
//
// Param flattening assumption: jax pytree order (nested dict keys sorted
// alphabetically). d_in[0..7] = x_v, edge_index_v, edge_attr_v, ptr_v,
// x_p, edge_index_p, edge_attr_p, ptr_p; d_in[8..153] = param leaves:
//   block1..block6 (21 leaves each: bn1.beta,bn1.gamma,bn2.beta,bn2.gamma,
//     bn_s.beta,bn_s.gamma, conv1.{bias,g,mu,root,sigma},
//     conv2.{...}, conv_s.{...}), then bn, bn_p, bn_v (beta,gamma),
//   conv_p, conv_v ({bias,g,mu,root,sigma}), fc1 (b,w), fc2 (b,w).
// =====================================================================

#define NN 16384      // nodes per modality
#define EE 131072     // edges per modality
#define BB 8          // graphs per batch

typedef __attribute__((ext_vector_type(16))) __bf16 v16bf;
typedef __attribute__((ext_vector_type(8)))  __bf16 v8bf;
typedef __attribute__((ext_vector_type(4)))  __bf16 v4bf;
typedef __attribute__((ext_vector_type(8)))  float  v8f;

// ---------------------------------------------------------------------
// Tiled bf16 WMMA GEMM: C[M,N] = A[M,K] (lda) * B[K,N] (ldb) (+ bias[n])
// 256 threads = 8 waves; 64x64 tile; K-step 32; each wave: 2 16x16 tiles
// sharing the B fragment -> 2x v_wmma_f32_16x16x32_bf16 per K-step.
// FAST path (all big GEMMs): M%64==0, N%64==0, K%32==0, 16B-aligned rows
// -> unconditional vector loads, no divergence, no per-load waits.
// ---------------------------------------------------------------------
template <bool FAST>
__global__ __launch_bounds__(256) void gemm_wmma_kernel(
    const float* __restrict__ A, int lda,
    const float* __restrict__ B, int ldb,
    float* __restrict__ C, int ldc,
    int M, int N, int K, const float* __restrict__ bias)
{
    __shared__ __align__(16) __bf16 As[64][48];  // row stride 96B (16B aligned)
    __shared__ __align__(16) __bf16 Bs[64][48];  // stored [n][k]

    const int tid  = threadIdx.x;
    const int lane = tid & 31;
    const int wave = tid >> 5;
    const int wm   = wave >> 2;   // 0..1 : row tiles wm*16 and wm*16+32
    const int wn   = wave & 3;    // 0..3 : col tile
    const int m0   = blockIdx.x * 64;
    const int n0   = blockIdx.y * 64;

    v8f acc0 = {};
    v8f acc1 = {};

    for (int kk = 0; kk < K; kk += 32) {
        if (FAST) {
            // ---- A: 64x32 floats = 512 float4; 2 per thread ----
            #pragma unroll
            for (int t = 0; t < 2; ++t) {
                int idx = tid + t * 256;
                int r = idx >> 3, c4 = idx & 7;
                float4 v = *(const float4*)(A + (long)(m0 + r) * lda + kk + c4 * 4);
                v4bf h = { (__bf16)v.x, (__bf16)v.y, (__bf16)v.z, (__bf16)v.w };
                *(v4bf*)&As[r][c4 * 4] = h;
            }
            // ---- B: read columns (coalesced across n), store transposed ----
            {
                int n = tid & 63, kq = tid >> 6;   // kq in 0..3 -> k = kq*8..kq*8+7
                const float* pb = B + (long)(kk + kq * 8) * ldb + n0 + n;
                v8bf h;
                #pragma unroll
                for (int j = 0; j < 8; ++j) { h[j] = (__bf16)pb[0]; pb += ldb; }
                *(v8bf*)&Bs[n][kq * 8] = h;
            }
        } else {
            for (int i = tid; i < 64 * 32; i += 256) {
                int r = i >> 5, c = i & 31;
                int gr = m0 + r, gc = kk + c;
                float v = (gr < M && gc < K) ? A[(long)gr * lda + gc] : 0.f;
                As[r][c] = (__bf16)v;
            }
            for (int i = tid; i < 32 * 64; i += 256) {
                int r = i >> 6, c = i & 63;        // r = k, c = n
                int gr = kk + r, gc = n0 + c;
                float v = (gr < K && gc < N) ? B[(long)gr * ldb + gc] : 0.f;
                Bs[c][r] = (__bf16)v;
            }
        }
        __syncthreads();

        const int ml = lane & 15;
        const int hi = lane >> 4;   // 0: lanes 0-15, 1: lanes 16-31
        // A fragment (ISA 7.12.2): lanes 0-15 hold K 0-7 & 16-23; lanes 16-31 hold K 8-15 & 24-31
        const v8bf* pa0 = (const v8bf*)&As[wm * 16 + ml][0];
        const v8bf* pa1 = (const v8bf*)&As[wm * 16 + 32 + ml][0];
        v8bf a0l = pa0[hi], a0h = pa0[hi + 2];
        v8bf a1l = pa1[hi], a1h = pa1[hi + 2];
        v16bf a0 = __builtin_shufflevector(a0l, a0h, 0,1,2,3,4,5,6,7,8,9,10,11,12,13,14,15);
        v16bf a1 = __builtin_shufflevector(a1l, a1h, 0,1,2,3,4,5,6,7,8,9,10,11,12,13,14,15);
        // B fragment: lanes 0-15 K 0-7 & 16-23; lanes 16-31 K 8-15 & 24-31 (column = lane%16)
        const v8bf* pb = (const v8bf*)&Bs[wn * 16 + ml][0];
        v8bf b0 = pb[hi], b1 = pb[hi + 2];
        v16bf b = __builtin_shufflevector(b0, b1, 0,1,2,3,4,5,6,7,8,9,10,11,12,13,14,15);

        acc0 = __builtin_amdgcn_wmma_f32_16x16x32_bf16(false, a0, false, b,
                                                       (short)0, acc0, false, false);
        acc1 = __builtin_amdgcn_wmma_f32_16x16x32_bf16(false, a1, false, b,
                                                       (short)0, acc1, false, false);
        __syncthreads();
    }

    // C/D layout: VGPR r -> M = r + (lane<16 ? 0 : 8); N = lane%16
    const int n = n0 + wn * 16 + (lane & 15);
    const int mb = m0 + wm * 16 + ((lane >> 4) << 3);
    if (FAST) {
        const float bv = bias ? bias[n] : 0.f;
        #pragma unroll
        for (int r = 0; r < 8; ++r) {
            C[(long)(mb + r) * ldc + n]      = acc0[r] + bv;
            C[(long)(mb + 32 + r) * ldc + n] = acc1[r] + bv;
        }
    } else if (n < N) {
        const float bv = bias ? bias[n] : 0.f;
        #pragma unroll
        for (int r = 0; r < 8; ++r) {
            int mm0 = mb + r;
            if (mm0 < M) C[(long)mm0 * ldc + n] = acc0[r] + bv;
            int mm1 = mb + 32 + r;
            if (mm1 < M) C[(long)mm1 * ldc + n] = acc1[r] + bv;
        }
    }
}

// --------------------- per-edge Gaussian mixture weights ---------------------
__global__ void edgew_kernel(const float* __restrict__ ea, const float* __restrict__ mu,
                             const float* __restrict__ sigma, float* __restrict__ wbuf,
                             int E, int Kk)
{
    int e = blockIdx.x * blockDim.x + threadIdx.x;
    if (e >= E) return;
    float a0 = ea[e * 3 + 0], a1 = ea[e * 3 + 1], a2 = ea[e * 3 + 2];
    for (int k = 0; k < Kk; ++k) {
        float d0 = a0 - mu[k * 3 + 0], d1 = a1 - mu[k * 3 + 1], d2 = a2 - mu[k * 3 + 2];
        float s0 = sigma[k * 3 + 0], s1 = sigma[k * 3 + 1], s2 = sigma[k * 3 + 2];
        float t = -0.5f * (d0 * d0 / (s0 * s0 + 1e-15f) +
                           d1 * d1 / (s1 * s1 + 1e-15f) +
                           d2 * d2 / (s2 * s2 + 1e-15f));
        wbuf[(long)e * Kk + k] = __expf(t);
    }
}

// --------------------- in-degree (float) ---------------------
__global__ void deg_kernel(const int* __restrict__ dst, float* __restrict__ deg, int E)
{
    int e = blockIdx.x * blockDim.x + threadIdx.x;
    if (e >= E) return;
    __hip_atomic_fetch_add(&deg[dst[e]], 1.0f, __ATOMIC_RELAXED, __HIP_MEMORY_SCOPE_AGENT);
}

// --------------------- weighted gather-scatter for one K slice ---------------------
__global__ void scatter_kernel(const float* __restrict__ xt, const float* __restrict__ wbuf,
                               int Kk, int k, const int* __restrict__ src,
                               const int* __restrict__ dst, float* __restrict__ agg, int cout)
{
    int e = blockIdx.x;
    int c = blockIdx.y * blockDim.x + threadIdx.x;
    if (c >= cout) return;
    float w = wbuf[(long)e * Kk + k];
    int s = src[e], d = dst[e];
    float v = xt[(long)s * cout + c] * w;
    __hip_atomic_fetch_add(&agg[(long)d * cout + c], v, __ATOMIC_RELAXED,
                           __HIP_MEMORY_SCOPE_AGENT);
}

// --------------------- out += agg / max(deg,1) ---------------------
__global__ void finalize_kernel(float* __restrict__ out, const float* __restrict__ agg,
                                const float* __restrict__ deg, int cout)
{
    int i = blockIdx.x;
    int c = blockIdx.y * blockDim.x + threadIdx.x;
    if (c >= cout) return;
    float dg = deg[i];
    dg = dg < 1.f ? 1.f : dg;
    out[(long)i * cout + c] += agg[(long)i * cout + c] / dg;
}

// --------------------- BN: per-channel sum / sumsq ---------------------
__global__ void bn_stats_kernel(const float* __restrict__ x, float* __restrict__ sums,
                                int n, int Cc)
{
    int c = blockIdx.x * blockDim.x + threadIdx.x;
    if (c >= Cc) return;
    int g = blockIdx.y, G = gridDim.y;
    float s = 0.f, s2 = 0.f;
    for (int i = g; i < n; i += G) {
        float v = x[(long)i * Cc + c];
        s += v; s2 += v * v;
    }
    __hip_atomic_fetch_add(&sums[c], s, __ATOMIC_RELAXED, __HIP_MEMORY_SCOPE_AGENT);
    __hip_atomic_fetch_add(&sums[Cc + c], s2, __ATOMIC_RELAXED, __HIP_MEMORY_SCOPE_AGENT);
}

// --------------------- BN apply (+optional residual, +optional ELU) ---------------------
__global__ void bn_apply_kernel(const float* __restrict__ x, const float* __restrict__ sums,
                                const float* __restrict__ gamma, const float* __restrict__ beta,
                                const float* __restrict__ residual, float* __restrict__ out,
                                int n, int Cc, int do_elu)
{
    int i = blockIdx.x;
    int c = blockIdx.y * blockDim.x + threadIdx.x;
    if (c >= Cc) return;
    float inv = 1.0f / (float)n;
    float mean = sums[c] * inv;
    float var = sums[Cc + c] * inv - mean * mean;
    float rstd = rsqrtf(var + 1e-5f);
    float v = gamma[c] * (x[(long)i * Cc + c] - mean) * rstd + beta[c];
    if (residual) v += residual[(long)i * Cc + c];
    if (do_elu) v = v > 0.f ? v : __expf(v) - 1.f;
    out[(long)i * Cc + c] = v;
}

// --------------------- head: gather virtual nodes, concat ---------------------
__global__ void gather_kernel(const float* __restrict__ hv, const float* __restrict__ hp,
                              const int* __restrict__ ptrv, const int* __restrict__ ptrp,
                              float* __restrict__ xcat)
{
    int b = blockIdx.x;
    int c = blockIdx.y * blockDim.x + threadIdx.x;
    if (c >= 1024) return;
    if (c < 512) xcat[b * 1024 + c] = hv[(long)(ptrv[b + 1] - 1) * 512 + c];
    else         xcat[b * 1024 + c] = hp[(long)(ptrp[b + 1] - 1) * 512 + (c - 512)];
}

// --------------------- small FC (+optional ELU) ---------------------
__global__ void fc_kernel(const float* __restrict__ in, const float* __restrict__ w,
                          const float* __restrict__ bias, float* __restrict__ out,
                          int ind, int outd, int do_elu)
{
    int b = blockIdx.x;
    int o = blockIdx.y * blockDim.x + threadIdx.x;
    if (o >= outd) return;
    float s = bias[o];
    for (int i = 0; i < ind; ++i) s += in[(long)b * ind + i] * w[(long)i * outd + o];
    if (do_elu) s = s > 0.f ? s : __expf(s) - 1.f;
    out[(long)b * outd + o] = s;
}

// --------------------- log_softmax over rows ---------------------
__global__ void lsm_kernel(const float* __restrict__ in, float* __restrict__ out, int Cc)
{
    __shared__ float red[512];
    int b = blockIdx.x, t = threadIdx.x;
    float v = (t < Cc) ? in[b * Cc + t] : -1e30f;
    red[t] = v; __syncthreads();
    for (int s = 256; s > 0; s >>= 1) { if (t < s) red[t] = fmaxf(red[t], red[t + s]); __syncthreads(); }
    float mx = red[0]; __syncthreads();
    float e = (t < Cc) ? __expf(v - mx) : 0.f;
    red[t] = e; __syncthreads();
    for (int s = 256; s > 0; s >>= 1) { if (t < s) red[t] += red[t + s]; __syncthreads(); }
    float lse = __logf(red[0]);
    if (t < Cc) out[b * Cc + t] = v - mx - lse;
}

// =====================================================================
// Host-side orchestration
// =====================================================================
struct GmmP { const float *g, *mu, *sigma, *root, *bias; int K; };

static inline void launch_gemm(hipStream_t st, const float* A, int lda, const float* B, int ldb,
                               float* C, int ldc, int M, int N, int K, const float* bias)
{
    dim3 grid((M + 63) / 64, (N + 63) / 64);
    bool fast = (M % 64 == 0) && (N % 64 == 0) && (K % 32 == 0) &&
                (lda % 4 == 0) && (ldb % 4 == 0);
    if (fast)
        gemm_wmma_kernel<true><<<grid, 256, 0, st>>>(A, lda, B, ldb, C, ldc, M, N, K, bias);
    else
        gemm_wmma_kernel<false><<<grid, 256, 0, st>>>(A, lda, B, ldb, C, ldc, M, N, K, bias);
}

static void run_conv(hipStream_t st, const float* x, int cin, int cout, const GmmP& p,
                     const int* src, const int* dst, const float* ea, const float* deg,
                     float* out, float* agg, float* xt, float* wbuf)
{
    edgew_kernel<<<(EE + 255) / 256, 256, 0, st>>>(ea, p.mu, p.sigma, wbuf, EE, p.K);
    // out = x @ root + bias
    launch_gemm(st, x, cin, p.root, cout, out, cout, NN, cout, cin, p.bias);
    hipMemsetAsync(agg, 0, (size_t)NN * cout * sizeof(float), st);
    for (int k = 0; k < p.K; ++k) {
        // xt = x @ g[:, k*cout:(k+1)*cout]
        launch_gemm(st, x, cin, p.g + (long)k * cout, p.K * cout, xt, cout, NN, cout, cin, nullptr);
        dim3 sg(EE, (cout + 255) / 256);
        scatter_kernel<<<sg, 256, 0, st>>>(xt, wbuf, p.K, k, src, dst, agg, cout);
    }
    dim3 fg(NN, (cout + 255) / 256);
    finalize_kernel<<<fg, 256, 0, st>>>(out, agg, deg, cout);
}

static void run_bn(hipStream_t st, const float* x, int Cc, const float* beta, const float* gamma,
                   const float* residual, int do_elu, float* sums, int nrows, float* outp)
{
    hipMemsetAsync(sums, 0, 2 * Cc * sizeof(float), st);
    bn_stats_kernel<<<dim3((Cc + 255) / 256, 64), 256, 0, st>>>(x, sums, nrows, Cc);
    bn_apply_kernel<<<dim3(nrows, (Cc + 255) / 256), 256, 0, st>>>(
        x, sums, gamma, beta, residual, outp, nrows, Cc, do_elu);
}

static void run_block(hipStream_t st, const float* const* P, int bb, int cin, int cout,
                      const int* src, const int* dst, const float* ea, const float* deg,
                      float* A, float* Bf, float* Cf, float* Df, float* G,
                      float* XT, float* wbuf, float* sums)
{
    // leaf layout within a block: bn1.{beta,gamma} bn2.{beta,gamma} bn_s.{beta,gamma}
    // conv1.{bias,g,mu,root,sigma} conv2.{...} conv_s.{...}
    GmmP c1{P[bb + 7],  P[bb + 8],  P[bb + 10], P[bb + 9],  P[bb + 6],  5};
    GmmP c2{P[bb + 12], P[bb + 13], P[bb + 15], P[bb + 14], P[bb + 11], 5};
    GmmP cs{P[bb + 17], P[bb + 18], P[bb + 20], P[bb + 19], P[bb + 16], 1};

    run_conv(st, A, cin, cout, c1, src, dst, ea, deg, Bf, G, XT, wbuf);       // h1 = conv1(x)
    run_bn(st, Bf, cout, P[bb + 0], P[bb + 1], nullptr, 1, sums, NN, Bf);     // h1 = elu(bn1(h1))
    run_conv(st, Bf, cout, cout, c2, src, dst, ea, deg, Cf, G, XT, wbuf);     // h2 = conv2(h1)
    run_conv(st, A, cin, cout, cs, src, dst, ea, deg, Df, G, XT, wbuf);       // s  = conv_s(x)
    run_bn(st, Df, cout, P[bb + 4], P[bb + 5], nullptr, 0, sums, NN, Df);     // s  = bn_s(s)
    run_bn(st, Cf, cout, P[bb + 2], P[bb + 3], Df, 1, sums, NN, A);           // out = elu(bn2(h2)+s)
}

extern "C" void kernel_launch(void* const* d_in, const int* in_sizes, int n_in,
                              void* d_out, int out_size, void* d_ws, size_t ws_size,
                              hipStream_t stream)
{
    (void)in_sizes; (void)out_size; (void)ws_size;

    const float* x_v  = (const float*)d_in[0];
    const int*   ei_v = (const int*)  d_in[1];
    const float* ea_v = (const float*)d_in[2];
    const int*   pt_v = (const int*)  d_in[3];
    const float* x_p  = (const float*)d_in[4];
    const int*   ei_p = (const int*)  d_in[5];
    const float* ea_p = (const float*)d_in[6];
    const int*   pt_p = (const int*)  d_in[7];

    const float* P[256];
    int nlim = n_in < 256 ? n_in : 256;
    for (int i = 0; i < nlim; ++i) P[i] = (const float*)d_in[i];

    // ---- workspace carving (~240 MB) ----
    char* wp = (char*)d_ws;
    auto alloc = [&](size_t bytes) {
        float* p = (float*)wp;
        wp += (bytes + 255) & ~(size_t)255;
        return p;
    };
    const size_t SLOT = (size_t)NN * 512 * sizeof(float);
    float* T0 = alloc(SLOT);  // branch-v activation (persists as h_v)
    float* T1 = alloc(SLOT);  // branch-p activation (persists as h_p)
    float* T2 = alloc(SLOT);
    float* T3 = alloc(SLOT);
    float* T4 = alloc(SLOT);  // agg scratch
    float* T5 = alloc(SLOT);
    float* XT = alloc(SLOT);  // one K-slice of x@g
    float* WB = alloc((size_t)EE * 5 * sizeof(float));
    float* DG = alloc((size_t)NN * sizeof(float));
    float* SM = alloc(2 * 512 * sizeof(float));
    float* XC = alloc((size_t)BB * 1024 * sizeof(float));
    float* F1 = alloc((size_t)BB * 512 * sizeof(float));
    float* LG = alloc((size_t)BB * 300 * sizeof(float));

    // ================= branch V =================
    const int* src_v = ei_v;
    const int* dst_v = ei_v + EE;
    hipMemsetAsync(DG, 0, (size_t)NN * sizeof(float), stream);
    deg_kernel<<<(EE + 255) / 256, 256, 0, stream>>>(dst_v, DG, EE);

    GmmP cv{P[146], P[147], P[149], P[148], P[145], 5};               // conv_v {g,mu,sigma,root,bias}
    run_conv(stream, x_v, 16, 64, cv, src_v, dst_v, ea_v, DG, T0, T4, XT, WB);
    run_bn(stream, T0, 64, P[138], P[139], nullptr, 1, SM, NN, T0);   // elu(bn_v(...))
    run_block(stream, P,  8,  64, 128, src_v, dst_v, ea_v, DG, T0, T2, T3, T5, T4, XT, WB, SM);
    run_block(stream, P, 29, 128, 256, src_v, dst_v, ea_v, DG, T0, T2, T3, T5, T4, XT, WB, SM);
    run_block(stream, P, 50, 256, 512, src_v, dst_v, ea_v, DG, T0, T2, T3, T5, T4, XT, WB, SM);
    // h_v in T0

    // ================= branch P =================
    const int* src_p = ei_p;
    const int* dst_p = ei_p + EE;
    hipMemsetAsync(DG, 0, (size_t)NN * sizeof(float), stream);
    deg_kernel<<<(EE + 255) / 256, 256, 0, stream>>>(dst_p, DG, EE);

    GmmP cp{P[141], P[142], P[144], P[143], P[140], 5};               // conv_p
    run_conv(stream, x_p, 1, 64, cp, src_p, dst_p, ea_p, DG, T1, T4, XT, WB);
    run_bn(stream, T1, 64, P[136], P[137], nullptr, 1, SM, NN, T1);   // elu(bn_p(...))
    run_block(stream, P,  71,  64, 128, src_p, dst_p, ea_p, DG, T1, T2, T3, T5, T4, XT, WB, SM);
    run_block(stream, P,  92, 128, 256, src_p, dst_p, ea_p, DG, T1, T2, T3, T5, T4, XT, WB, SM);
    run_block(stream, P, 113, 256, 512, src_p, dst_p, ea_p, DG, T1, T2, T3, T5, T4, XT, WB, SM);
    // h_p in T1

    // ================= head =================
    gather_kernel<<<dim3(BB, 4), 256, 0, stream>>>(T0, T1, pt_v, pt_p, XC);
    fc_kernel<<<dim3(BB, 2), 256, 0, stream>>>(XC, P[151], P[150], F1, 1024, 512, 1); // elu(fc1)
    run_bn(stream, F1, 512, P[134], P[135], nullptr, 0, SM, BB, F1);                  // bn (train)
    fc_kernel<<<dim3(BB, 2), 256, 0, stream>>>(F1, P[153], P[152], LG, 512, 300, 0);  // fc2
    lsm_kernel<<<BB, 512, 0, stream>>>(LG, (float*)d_out, 300);
}
